// CPImplicitPairwise_23252952941210
// MI455X (gfx1250) — compile-verified, matrix operand order checked by reading, not run on hardware
//
#include <hip/hip_runtime.h>
#include <stdint.h>

// ------------------------------------------------------------------
// CDNA5 (gfx1250) wave32 WMMA types
// ------------------------------------------------------------------
typedef __bf16 bf16;
typedef __attribute__((ext_vector_type(16))) __bf16 bf16x16;
typedef __attribute__((ext_vector_type(8)))  float  f32x8;

static __device__ __forceinline__ f32x8 wmma_bf16(bf16x16 a, bf16x16 b, f32x8 c) {
  // v_wmma_f32_16x16x32_bf16  D = A(16x32) * B(32x16) + C
  return __builtin_amdgcn_wmma_f32_16x16x32_bf16(false, a, false, b, (short)0, c, false, false);
}

static __device__ __forceinline__ float u01(uint32_t x) {
  x ^= x >> 17; x *= 0xed5ad4bbu; x ^= x >> 11; x *= 0xac4c1b51u;
  x ^= x >> 15; x *= 0x31848babu; x ^= x >> 14;
  return ((float)(x >> 8) + 0.5f) * (1.0f / 16777216.0f);
}

// ------------------------------------------------------------------
// Generic 16-row x 256-col WMMA MLP layer.  One wave owns one 16-row
// tile (wave id = row tile), loops over the 16 column tiles.
// A fragment (16x32 bf16): lane<16 -> row=lane, K=k0..k0+15
//                          lane>=16 -> row=lane-16, K=k0+16..k0+31
// B fragment comes pre-packed in exactly the lane layout (32B/lane).
// C/D f32 layout: vgpr i -> M = i + (lane<16?0:8), N = lane%16.
// ------------------------------------------------------------------
static __device__ __forceinline__ void mlp_layer_wave(
    int wave, int lane,
    const bf16* __restrict__ inBuf, int inStride,
    bf16* __restrict__ outBuf, int outStride,
    int ksteps, const uint32_t* __restrict__ wpk,
    const float* __restrict__ bias, bool doRelu)
{
  const int lm   = lane & 15;
  const int koff = (lane & 16) ? 16 : 0;
  const bf16* inRow = inBuf + (wave * 16 + lm) * inStride + koff;
  for (int j = 0; j < 16; ++j) {
    if (j < 15)  // stream next column tile of weights through L2
      __builtin_prefetch(wpk + (size_t)(((j + 1) * ksteps) * 32 + lane) * 8, 0, 1);
    f32x8 acc = {0.f, 0.f, 0.f, 0.f, 0.f, 0.f, 0.f, 0.f};
    for (int s = 0; s < ksteps; ++s) {
      bf16x16 a = *(const bf16x16*)(inRow + s * 32);
      bf16x16 b = *(const bf16x16*)(wpk + (size_t)((j * ksteps + s) * 32 + lane) * 8);
      acc = wmma_bf16(a, b, acc);
    }
    const int   col = j * 16 + lm;
    const float bv  = bias[col];
#pragma unroll
    for (int i = 0; i < 8; ++i) {
      float x = acc[i] + bv;
      if (doRelu) x = fmaxf(x, 0.0f);
      const int mrow = i + ((lane & 16) ? 8 : 0);
      outBuf[(wave * 16 + mrow) * outStride + col] = (bf16)x;
    }
  }
}

// ------------------------------------------------------------------
// Pack fp32 weight matrix W[Fout, rowStride] (slice starting at colOff,
// Fin columns) into bf16 WMMA-B tiles: dword index
//   ((j*ksteps + s)*32 + lane)*8 + t  holds B[k0+2t .. +1][n0+lane%16]
// ------------------------------------------------------------------
__global__ void __launch_bounds__(256) pack_w_k(const float* __restrict__ W,
                                                uint32_t* __restrict__ out,
                                                int Fout, int Fin, int colOff, int rowStride) {
  int idx = blockIdx.x * 256 + threadIdx.x;
  int total = (Fout >> 4) * (Fin >> 5) * 256;
  if (idx >= total) return;
  int t    = idx & 7;
  int lane = (idx >> 3) & 31;
  int rest = idx >> 8;
  int ksteps = Fin >> 5;
  int s = rest % ksteps;
  int j = rest / ksteps;
  int n = j * 16 + (lane & 15);
  int k = s * 32 + ((lane & 16) ? 16 : 0) + 2 * t;
  union { bf16 h[2]; uint32_t u; } p;
  p.h[0] = (bf16)W[(size_t)n * rowStride + colOff + k];
  p.h[1] = (bf16)W[(size_t)n * rowStride + colOff + k + 1];
  out[idx] = p.u;
}

// out[b, 0:64, :] = x_in ; out[b, 64:384, :] = 0
__global__ void __launch_bounds__(256) base_k(const float* __restrict__ x_in,
                                              float* __restrict__ out) {
  size_t i = (size_t)blockIdx.x * 256 + threadIdx.x;
  if (i >= (size_t)2 * 384 * 8192) return;
  const size_t per = (size_t)384 * 8192;
  int b = (int)(i / per);
  size_t rem = i - (size_t)b * per;
  int ch  = (int)(rem >> 13);
  int pos = (int)(rem & 8191);
  out[i] = (ch < 64) ? x_in[(((size_t)b * 64 + ch) << 13) + pos] : 0.0f;
}

// xi[b][n][c] = x_in[b][c][sel[b][n]] (bf16), coords[b][n][d] gather
__global__ void __launch_bounds__(256) gather_k(const float* __restrict__ x_in,
                                                const int* __restrict__ sel,
                                                const float* __restrict__ pts,
                                                bf16* __restrict__ XI,
                                                float* __restrict__ COORDS) {
  int i = blockIdx.x * 256 + threadIdx.x;
  if (i < 2 * 1024 * 64) {
    int c = i & 63, n = (i >> 6) & 1023, b = i >> 16;
    int idx = sel[(b << 10) + n];
    XI[i] = (bf16)x_in[(((size_t)((b << 6) + c)) << 13) + idx];
  }
  if (i < 2 * 1024 * 3) {
    int d = i % 3, nn = i / 3;
    int b = nn >> 10, n = nn & 1023;
    int idx = sel[(b << 10) + n];
    COORDS[i] = pts[(((size_t)b << 13) + idx) * 3 + d];
  }
}

// ------------------------------------------------------------------
// relate MLP: rows = 2048 (b,n).  128 rows/block, 8 waves.
// L0(64->256) L1(256->256) L2(288->256 concat c) L3(256->256) out->sigmoid
// ------------------------------------------------------------------
__global__ void __launch_bounds__(256) relate_k(
    const bf16* __restrict__ XI, const float* __restrict__ COORDS,
    const uint32_t* __restrict__ L0pk, const float* __restrict__ L0b,
    const uint32_t* __restrict__ L1pk, const float* __restrict__ L1b,
    const uint32_t* __restrict__ L2pk, const float* __restrict__ L2b,
    const uint32_t* __restrict__ L3pk, const float* __restrict__ L3b,
    const float* __restrict__ outW, const float* __restrict__ outB,
    const float* __restrict__ p0w, const float* __restrict__ p0b,
    const float* __restrict__ p1w, const float* __restrict__ p1b,
    float* __restrict__ RP) {
  __shared__ alignas(32) bf16 buf0[128 * 288];
  __shared__ alignas(32) bf16 buf1[128 * 288];
  const int tid = threadIdx.x, wave = tid >> 5, lane = tid & 31;
  const int rowBase = blockIdx.x * 128;
  // load xi rows (64 bf16 = 32 dwords each)
  const uint32_t* src = (const uint32_t*)XI + (size_t)rowBase * 32;
  for (int it = tid; it < 128 * 32; it += 256) {
    int r = it >> 5, dc = it & 31;
    ((uint32_t*)(&buf0[r * 288]))[dc] = src[it];
  }
  // coord path c = relu(proj1(proj0(cr))) -> cols 256..287
  if (tid < 128) {
    int g = rowBase + tid;
    float c0 = COORDS[g * 3], c1 = COORDS[g * 3 + 1], c2 = COORDS[g * 3 + 2];
    float p0[32];
#pragma unroll
    for (int f = 0; f < 32; ++f)
      p0[f] = p0w[f * 3] * c0 + p0w[f * 3 + 1] * c1 + p0w[f * 3 + 2] * c2 + p0b[f];
    for (int f = 0; f < 32; ++f) {
      float a = p1b[f];
#pragma unroll
      for (int jj = 0; jj < 32; ++jj) a += p1w[f * 32 + jj] * p0[jj];
      buf0[tid * 288 + 256 + f] = (bf16)fmaxf(a, 0.0f);
    }
  }
  __syncthreads();
  mlp_layer_wave(wave, lane, buf0, 288, buf1, 288, 2, L0pk, L0b, true);
  __syncthreads();
  mlp_layer_wave(wave, lane, buf1, 288, buf0, 288, 8, L1pk, L1b, true);  // c stays in cols 256..287
  __syncthreads();
  mlp_layer_wave(wave, lane, buf0, 288, buf1, 288, 9, L2pk, L2b, true);  // K = 288
  __syncthreads();
  mlp_layer_wave(wave, lane, buf1, 288, buf0, 288, 8, L3pk, L3b, true);
  __syncthreads();
  if (tid < 128) {
    float a = outB[0];
    for (int kk = 0; kk < 256; ++kk) a += (float)buf0[tid * 288 + kk] * outW[kk];
    RP[rowBase + tid] = 1.0f / (1.0f + __expf(-a));
  }
}

// ------------------------------------------------------------------
// Categorical sampling (inverse CDF, hashed uniforms) + pack topk_feat
// into WMMA-B tiles for the aggregation GEMM.
// ------------------------------------------------------------------
__global__ void __launch_bounds__(1024) sample_k(const float* __restrict__ RP,
                                                 int* __restrict__ TOPK,
                                                 float* __restrict__ RPK,
                                                 const bf16* __restrict__ XI,
                                                 uint32_t* __restrict__ FEATPK) {
  __shared__ float sc[1024];
  __shared__ int   tk[64];
  const int b = blockIdx.x, tid = threadIdx.x;
  sc[tid] = RP[(b << 10) + tid];
  __syncthreads();
  for (int off = 1; off < 1024; off <<= 1) {   // Hillis-Steele inclusive scan
    float add = (tid >= off) ? sc[tid - off] : 0.0f;
    __syncthreads();
    sc[tid] += add;
    __syncthreads();
  }
  const float total = sc[1023];
  if (tid < 64) {
    float u = u01((uint32_t)(b * 64 + tid) * 2654435761u + 0x9e3779b9u);
    float target = u * total;
    int lo = 0, hi = 1023;
    while (lo < hi) { int mid = (lo + hi) >> 1; if (sc[mid] > target) hi = mid; else lo = mid + 1; }
    tk[tid] = lo;
    TOPK[b * 64 + tid] = lo;
    RPK[b * 64 + tid]  = RP[(b << 10) + lo];
  }
  __syncthreads();
  // pack feat B tiles: B[k=0..63][c=0..63], 4 col tiles x 2 k-steps
  for (int it = tid; it < 2048; it += 1024) {
    int t = it & 7, lane = (it >> 3) & 31, rest = it >> 8;
    int s = rest & 1, j = rest >> 1;
    int c = j * 16 + (lane & 15);
    int k = s * 32 + ((lane & 16) ? 16 : 0) + 2 * t;
    union { bf16 h[2]; uint32_t u; } p;
    p.h[0] = XI[((b << 10) + tk[k]) * 64 + c];
    p.h[1] = XI[((b << 10) + tk[k + 1]) * 64 + c];
    FEATPK[b * 2048 + it] = p.u;
  }
}

// ------------------------------------------------------------------
// Fold proj0/proj1 of the P-MLP:  c(n,k) = relu(u[n] + v[k]).
// u[n] = M1*cr_n + u0 ,  v[k] = M2*ct_k ,  M1 = W1(A-B), M2 = W1*B
// ------------------------------------------------------------------
__global__ void __launch_bounds__(256) uv_k(const float* __restrict__ p0w,
                                            const float* __restrict__ p0b,
                                            const float* __restrict__ p1w,
                                            const float* __restrict__ p1b,
                                            const float* __restrict__ COORDS,
                                            const int* __restrict__ TOPK,
                                            float* __restrict__ Uo,
                                            float* __restrict__ Vo) {
  __shared__ float M1[64 * 3], M2[64 * 3], U0[64];
  const int b = blockIdx.x, tid = threadIdx.x;
  if (tid < 64) {
    int f = tid;
    float u0 = p1b[f], m1[3] = {0, 0, 0}, m2[3] = {0, 0, 0};
    for (int jj = 0; jj < 64; ++jj) {
      float w1 = p1w[f * 64 + jj];
      u0 += w1 * p0b[jj];
#pragma unroll
      for (int d = 0; d < 3; ++d) {
        float a = p0w[jj * 6 + d], c2 = p0w[jj * 6 + 3 + d];
        m1[d] += w1 * (a - c2);
        m2[d] += w1 * c2;
      }
    }
    U0[f] = u0;
#pragma unroll
    for (int d = 0; d < 3; ++d) { M1[f * 3 + d] = m1[d]; M2[f * 3 + d] = m2[d]; }
  }
  __syncthreads();
  for (int n = tid; n < 1024; n += 256) {
    const float* cr = COORDS + ((size_t)(b << 10) + n) * 3;
    float c0 = cr[0], c1 = cr[1], c2 = cr[2];
    for (int f = 0; f < 64; ++f)
      Uo[(((size_t)b << 10) + n) * 64 + f] =
          M1[f * 3] * c0 + M1[f * 3 + 1] * c1 + M1[f * 3 + 2] * c2 + U0[f];
  }
  if (tid < 64) {
    int idx = TOPK[b * 64 + tid];
    const float* ct = COORDS + ((size_t)(b << 10) + idx) * 3;
    float c0 = ct[0], c1 = ct[1], c2 = ct[2];
    for (int f = 0; f < 64; ++f)
      Vo[(b * 64 + tid) * 64 + f] = M2[f * 3] * c0 + M2[f * 3 + 1] * c1 + M2[f * 3 + 2] * c2;
  }
}

// ------------------------------------------------------------------
// P-MLP front (per-n only): L0,L1,L2 relu, then t3 = W3a*h2 + b3 (f32)
// ------------------------------------------------------------------
__global__ void __launch_bounds__(256) hpath_k(
    const bf16* __restrict__ XI,
    const uint32_t* __restrict__ L0pk, const float* __restrict__ L0b,
    const uint32_t* __restrict__ L1pk, const float* __restrict__ L1b,
    const uint32_t* __restrict__ L2pk, const float* __restrict__ L2b,
    const uint32_t* __restrict__ W3Apk, const float* __restrict__ b3,
    float* __restrict__ T3) {
  __shared__ alignas(32) bf16 buf0[128 * 256];
  __shared__ alignas(32) bf16 buf1[128 * 256];
  const int tid = threadIdx.x, wave = tid >> 5, lane = tid & 31;
  const int rowBase = blockIdx.x * 128;
  const uint32_t* src = (const uint32_t*)XI + (size_t)rowBase * 32;
  for (int it = tid; it < 128 * 32; it += 256) {
    int r = it >> 5, dc = it & 31;
    ((uint32_t*)(&buf0[r * 256]))[dc] = src[it];
  }
  __syncthreads();
  mlp_layer_wave(wave, lane, buf0, 256, buf1, 256, 2, L0pk, L0b, true);
  __syncthreads();
  mlp_layer_wave(wave, lane, buf1, 256, buf0, 256, 8, L1pk, L1b, true);
  __syncthreads();
  mlp_layer_wave(wave, lane, buf0, 256, buf1, 256, 8, L2pk, L2b, true);
  __syncthreads();
  // t3 = W3a @ h2 + b3, f32 straight from fragments (no relu)
  const int lm = lane & 15, koff = (lane & 16) ? 16 : 0;
  const bf16* inRow = buf1 + (wave * 16 + lm) * 256 + koff;
  for (int j = 0; j < 16; ++j) {
    f32x8 acc = {0.f, 0.f, 0.f, 0.f, 0.f, 0.f, 0.f, 0.f};
    for (int s = 0; s < 8; ++s) {
      bf16x16 a = *(const bf16x16*)(inRow + s * 32);
      bf16x16 bw = *(const bf16x16*)(W3Apk + (size_t)((j * 8 + s) * 32 + lane) * 8);
      acc = wmma_bf16(a, bw, acc);
    }
    const int col = j * 16 + lm;
    const float bv = b3[col];
#pragma unroll
    for (int i = 0; i < 8; ++i) {
      int mrow = i + ((lane & 16) ? 8 : 0);
      T3[((size_t)(rowBase + wave * 16 + mrow)) * 256 + col] = acc[i] + bv;
    }
  }
}

// ------------------------------------------------------------------
// Main heavy kernel: per block a (16n x 8k) row tile (128 rows).
// rows r = klocal*16 + nlocal, wave = klocal (row tile has fixed k).
// layer3(K=64,+t3), L4, L5 (K=256) WMMA; head+softmax*rp_k scalar.
// ------------------------------------------------------------------
__global__ void __launch_bounds__(256) pmlp_main_k(
    const float* __restrict__ U, const float* __restrict__ V,
    const float* __restrict__ T3, const float* __restrict__ RPK,
    const uint32_t* __restrict__ W3Bpk,
    const uint32_t* __restrict__ L4pk, const float* __restrict__ b4,
    const uint32_t* __restrict__ L5pk, const float* __restrict__ b5,
    const float* __restrict__ outW, const float* __restrict__ outB,
    bf16* __restrict__ PW) {
  __shared__ alignas(32) bf16 bufC[128 * 64];
  __shared__ alignas(32) bf16 bufA[128 * 256];
  __shared__ alignas(32) bf16 bufB[128 * 256];
  const int tid = threadIdx.x, wave = tid >> 5, lane = tid & 31;
  const int blk = blockIdx.x;
  const int kblk = blk & 7, ntile = (blk >> 3) & 63, b = blk >> 9;
  {  // c = relu(u[n] + v[k])
    int r = tid & 127, half = tid >> 7;
    int n = ntile * 16 + (r & 15), k = kblk * 8 + (r >> 4);
    const float* up = U + (((size_t)(b << 10) + n) * 64) + half * 32;
    const float* vp = V + ((size_t)(b * 64 + k) * 64) + half * 32;
#pragma unroll
    for (int f = 0; f < 32; ++f)
      bufC[r * 64 + half * 32 + f] = (bf16)fmaxf(up[f] + vp[f], 0.0f);
  }
  __syncthreads();
  {  // layer3: relu(t3[n] + W3b @ c)
    const int lm = lane & 15, koff = (lane & 16) ? 16 : 0;
    const bf16* inRow = bufC + (wave * 16 + lm) * 64 + koff;
    for (int j = 0; j < 16; ++j) {
      f32x8 acc = {0.f, 0.f, 0.f, 0.f, 0.f, 0.f, 0.f, 0.f};
      for (int s = 0; s < 2; ++s) {
        bf16x16 a = *(const bf16x16*)(inRow + s * 32);
        bf16x16 bw = *(const bf16x16*)(W3Bpk + (size_t)((j * 2 + s) * 32 + lane) * 8);
        acc = wmma_bf16(a, bw, acc);
      }
      const int col = j * 16 + lm;
#pragma unroll
      for (int i = 0; i < 8; ++i) {
        int mrow = i + ((lane & 16) ? 8 : 0);
        int n = ntile * 16 + mrow;
        float x = acc[i] + T3[(((size_t)(b << 10)) + n) * 256 + col];
        bufA[(wave * 16 + mrow) * 256 + col] = (bf16)fmaxf(x, 0.0f);
      }
    }
  }
  __syncthreads();
  mlp_layer_wave(wave, lane, bufA, 256, bufB, 256, 8, L4pk, b4, true);
  __syncthreads();
  mlp_layer_wave(wave, lane, bufB, 256, bufA, 256, 8, L5pk, b5, true);
  __syncthreads();
  if (tid < 128) {  // head: 5 logits, softmax, * rp_k
    int r = tid;
    int n = ntile * 16 + (r & 15), k = kblk * 8 + (r >> 4);
    float lg[5];
    for (int rel = 0; rel < 5; ++rel) {
      float a = outB[rel];
      const float* wr = outW + rel * 256;
      for (int kk = 0; kk < 256; ++kk) a += (float)bufA[r * 256 + kk] * wr[kk];
      lg[rel] = a;
    }
    float mx = lg[0];
#pragma unroll
    for (int rel = 1; rel < 5; ++rel) mx = fmaxf(mx, lg[rel]);
    float se = 0.0f;
#pragma unroll
    for (int rel = 0; rel < 5; ++rel) { lg[rel] = __expf(lg[rel] - mx); se += lg[rel]; }
    float scale = RPK[b * 64 + k] / se;
#pragma unroll
    for (int rel = 0; rel < 5; ++rel)
      PW[(((size_t)((b * 5 + rel) << 10)) + n) * 64 + k] = (bf16)(lg[rel] * scale);
  }
}

// ------------------------------------------------------------------
// ctx[rel][n][c] = (P_w[rel][n][:] @ feat[:,c]) / sum_k P_w, scattered
// into out channels 64 + rel*64 + c at column sel[b][n].  WMMA GEMM.
// ------------------------------------------------------------------
__global__ void __launch_bounds__(256) agg_k(const bf16* __restrict__ PW,
                                             const uint32_t* __restrict__ FEATPK,
                                             const int* __restrict__ sel,
                                             float* __restrict__ out) {
  __shared__ float denom[80];  // [rel][nlocal]
  const int blk = blockIdx.x, ntile = blk & 63, b = blk >> 6;
  const int tid = threadIdx.x, wave = tid >> 5, lane = tid & 31;
  if (tid < 80) {
    int rel = tid / 16, nloc = tid & 15;
    const bf16* p = PW + (((size_t)((b * 5 + rel) << 10)) + ntile * 16 + nloc) * 64;
    float s = 0.0f;
    for (int k = 0; k < 64; ++k) s += (float)p[k];
    denom[tid] = s;
  }
  __syncthreads();
  const int lm = lane & 15, koff = (lane & 16) ? 16 : 0;
  for (int u = wave; u < 20; u += 8) {  // 5 rel x 4 col tiles
    int rel = u % 5, j = u / 5;
    f32x8 acc = {0.f, 0.f, 0.f, 0.f, 0.f, 0.f, 0.f, 0.f};
    for (int s = 0; s < 2; ++s) {
      const bf16* ap = PW + (((size_t)((b * 5 + rel) << 10)) + ntile * 16 + lm) * 64 + s * 32 + koff;
      bf16x16 a = *(const bf16x16*)ap;
      bf16x16 bw = *(const bf16x16*)(FEATPK + b * 2048 + (size_t)((j * 2 + s) * 32 + lane) * 8);
      acc = wmma_bf16(a, bw, acc);
    }
#pragma unroll
    for (int i = 0; i < 8; ++i) {
      int mrow = i + ((lane & 16) ? 8 : 0);
      int n = ntile * 16 + mrow;
      float val = acc[i] / denom[rel * 16 + mrow];
      int col = sel[(b << 10) + n];
      int ch = 64 + rel * 64 + j * 16 + lm;
      out[(((size_t)(b * 384 + ch)) << 13) + col] = val;
    }
  }
}

// ==================================================================
extern "C" void kernel_launch(void* const* d_in, const int* in_sizes, int n_in,
                              void* d_out, int out_size, void* d_ws, size_t ws_size,
                              hipStream_t stream) {
  (void)in_sizes; (void)n_in; (void)out_size;
  const float* x_in = (const float*)d_in[0];
  const int*   sel  = (const int*)d_in[1];
  const float* pts  = (const float*)d_in[2];
  // params flattened (jax tree order: "P" < "relate"; per-MLP: layers, out, proj0, proj1; per-lin: b, w)
  const float* P_l0b = (const float*)d_in[4];  const float* P_l0w = (const float*)d_in[5];
  const float* P_l1b = (const float*)d_in[6];  const float* P_l1w = (const float*)d_in[7];
  const float* P_l2b = (const float*)d_in[8];  const float* P_l2w = (const float*)d_in[9];
  const float* P_l3b = (const float*)d_in[10]; const float* P_l3w = (const float*)d_in[11];
  const float* P_l4b = (const float*)d_in[12]; const float* P_l4w = (const float*)d_in[13];
  const float* P_l5b = (const float*)d_in[14]; const float* P_l5w = (const float*)d_in[15];
  const float* P_ob  = (const float*)d_in[16]; const float* P_ow  = (const float*)d_in[17];
  const float* P_p0b = (const float*)d_in[18]; const float* P_p0w = (const float*)d_in[19];
  const float* P_p1b = (const float*)d_in[20]; const float* P_p1w = (const float*)d_in[21];
  const float* R_l0b = (const float*)d_in[22]; const float* R_l0w = (const float*)d_in[23];
  const float* R_l1b = (const float*)d_in[24]; const float* R_l1w = (const float*)d_in[25];
  const float* R_l2b = (const float*)d_in[26]; const float* R_l2w = (const float*)d_in[27];
  const float* R_l3b = (const float*)d_in[28]; const float* R_l3w = (const float*)d_in[29];
  const float* R_ob  = (const float*)d_in[30]; const float* R_ow  = (const float*)d_in[31];
  const float* R_p0b = (const float*)d_in[32]; const float* R_p0w = (const float*)d_in[33];
  const float* R_p1b = (const float*)d_in[34]; const float* R_p1w = (const float*)d_in[35];
  float* out = (float*)d_out;

  // workspace carve-up (all sizes multiples of 256B)
  size_t off = 0;
  char* wsb = (char*)d_ws;
  auto take = [&](size_t bytes) -> char* { char* p = wsb + off; off += bytes; return p; };
  bf16*     XI     = (bf16*)    take((size_t)2 * 1024 * 64 * 2);
  float*    COORDS = (float*)   take((size_t)2 * 1024 * 3 * 4);
  float*    RP     = (float*)   take((size_t)2 * 1024 * 4);
  int*      TOPK   = (int*)     take((size_t)2 * 64 * 4);
  float*    RPK    = (float*)   take((size_t)2 * 64 * 4);
  uint32_t* FEATPK = (uint32_t*)take((size_t)2 * 2048 * 4);
  float*    Uo     = (float*)   take((size_t)2 * 1024 * 64 * 4);
  float*    Vo     = (float*)   take((size_t)2 * 64 * 64 * 4);
  float*    T3     = (float*)   take((size_t)2 * 1024 * 256 * 4);
  bf16*     PW     = (bf16*)    take((size_t)2 * 5 * 1024 * 64 * 2);
  uint32_t* RL0PK  = (uint32_t*)take((size_t)256 * 64 * 2);
  uint32_t* RL1PK  = (uint32_t*)take((size_t)256 * 256 * 2);
  uint32_t* RL2PK  = (uint32_t*)take((size_t)256 * 288 * 2);
  uint32_t* RL3PK  = (uint32_t*)take((size_t)256 * 256 * 2);
  uint32_t* PL0PK  = (uint32_t*)take((size_t)256 * 64 * 2);
  uint32_t* PL1PK  = (uint32_t*)take((size_t)256 * 256 * 2);
  uint32_t* PL2PK  = (uint32_t*)take((size_t)256 * 256 * 2);
  uint32_t* W3APK  = (uint32_t*)take((size_t)256 * 256 * 2);
  uint32_t* W3BPK  = (uint32_t*)take((size_t)256 * 64 * 2);
  uint32_t* PL4PK  = (uint32_t*)take((size_t)256 * 256 * 2);
  uint32_t* PL5PK  = (uint32_t*)take((size_t)256 * 256 * 2);
  if (off > ws_size) return;

  // 1) pack all GEMM weights into WMMA-B tiles  (blocks = (Fout/16)*(Fin/32))
  pack_w_k<<<dim3(16 * 2), 256, 0, stream>>>(R_l0w, RL0PK, 256, 64, 0, 64);
  pack_w_k<<<dim3(16 * 8), 256, 0, stream>>>(R_l1w, RL1PK, 256, 256, 0, 256);
  pack_w_k<<<dim3(16 * 9), 256, 0, stream>>>(R_l2w, RL2PK, 256, 288, 0, 288);
  pack_w_k<<<dim3(16 * 8), 256, 0, stream>>>(R_l3w, RL3PK, 256, 256, 0, 256);
  pack_w_k<<<dim3(16 * 2), 256, 0, stream>>>(P_l0w, PL0PK, 256, 64, 0, 64);
  pack_w_k<<<dim3(16 * 8), 256, 0, stream>>>(P_l1w, PL1PK, 256, 256, 0, 256);
  pack_w_k<<<dim3(16 * 8), 256, 0, stream>>>(P_l2w, PL2PK, 256, 256, 0, 256);
  pack_w_k<<<dim3(16 * 8), 256, 0, stream>>>(P_l3w, W3APK, 256, 256, 0, 320);
  pack_w_k<<<dim3(16 * 2), 256, 0, stream>>>(P_l3w, W3BPK, 256, 64, 256, 320);
  pack_w_k<<<dim3(16 * 8), 256, 0, stream>>>(P_l4w, PL4PK, 256, 256, 0, 256);
  pack_w_k<<<dim3(16 * 8), 256, 0, stream>>>(P_l5w, PL5PK, 256, 256, 0, 256);
  // 2) output init (x_in passthrough + zero ctx channels)
  base_k<<<dim3(24576), 256, 0, stream>>>(x_in, out);
  // 3) gathers
  gather_k<<<dim3(512), 256, 0, stream>>>(x_in, sel, pts, XI, COORDS);
  // 4) relate MLP -> rp
  relate_k<<<dim3(16), 256, 0, stream>>>(XI, COORDS, RL0PK, R_l0b, RL1PK, R_l1b,
                                         RL2PK, R_l2b, RL3PK, R_l3b,
                                         R_ow, R_ob, R_p0w, R_p0b, R_p1w, R_p1b, RP);
  // 5) categorical sampling + feat B-tile packing
  sample_k<<<dim3(2), 1024, 0, stream>>>(RP, TOPK, RPK, XI, FEATPK);
  // 6) folded proj path  u[n], v[k]
  uv_k<<<dim3(2), 256, 0, stream>>>(P_p0w, P_p0b, P_p1w, P_p1b, COORDS, TOPK, Uo, Vo);
  // 7) per-n P-MLP front  t3 = W3a@h2 + b3
  hpath_k<<<dim3(16), 256, 0, stream>>>(XI, PL0PK, P_l0b, PL1PK, P_l1b,
                                        PL2PK, P_l2b, W3APK, P_l3b, T3);
  // 8) heavy per-(n,k) layers + softmax head
  pmlp_main_k<<<dim3(1024), 256, 0, stream>>>(Uo, Vo, T3, RPK, W3BPK,
                                              PL4PK, P_l4b, PL5PK, P_l5b,
                                              P_ow, P_ob, PW);
  // 9) aggregation GEMM + scatter
  agg_k<<<dim3(128), 256, 0, stream>>>(PW, FEATPK, sel, out);
}
